// ProteinFeatures_51548197486904
// MI455X (gfx1250) — matrix operand order repaired and checked by reading, not compiled, not run on hardware
//
#include <hip/hip_runtime.h>
#include <hip/hip_bf16.h>

#define BB   4
#define LL   2048
#define KNB  30
#define NF   128
#define EIN  55
#define EINP 64

typedef __attribute__((ext_vector_type(16))) _Float16 v16h;
typedef __attribute__((ext_vector_type(8)))  float    v8f;

static __device__ __forceinline__ float sgnf(float x) {
    return (x > 0.f) ? 1.f : ((x < 0.f) ? -1.f : 0.f);
}

// ---------------------------------------------------------------------------
// Kernel 1: masked distance row + top-30 selection (stable, smallest index on
// ties via 64-bit key = float_bits<<32 | index; distances are >= 0).
// One block of 256 threads per (b,i) row.
// ---------------------------------------------------------------------------
__global__ __launch_bounds__(256) void pf_topk_kernel(
    const float* __restrict__ X, const float* __restrict__ mask,
    float* __restrict__ d_nb, int* __restrict__ e_idx,
    float* __restrict__ e_idx_f) {
  const int row = blockIdx.x;          // b*L + i
  const int b   = row / LL;
  const int tid = threadIdx.x;

  __shared__ unsigned long long s_key[LL];
  __shared__ unsigned long long s_red[256];
  __shared__ float s_fred[256];

  const float xi = X[(row * 4 + 1) * 3 + 0];
  const float yi = X[(row * 4 + 1) * 3 + 1];
  const float zi = X[(row * 4 + 1) * 3 + 2];
  const float mi = mask[row];

  float dm[8], m2[8];
  float lmax = 0.f;
#pragma unroll
  for (int s = 0; s < 8; ++s) {
    const int j  = tid + s * 256;
    const int gj = b * LL + j;
    const float dx = X[(gj * 4 + 1) * 3 + 0] - xi;
    const float dy = X[(gj * 4 + 1) * 3 + 1] - yi;
    const float dz = X[(gj * 4 + 1) * 3 + 2] - zi;
    const float m2d = mi * mask[gj];
    const float d = m2d * sqrtf(dx * dx + dy * dy + dz * dz + 1e-6f);
    dm[s] = d; m2[s] = m2d;
    lmax = fmaxf(lmax, d);
  }
  s_fred[tid] = lmax; __syncthreads();
  for (int off = 128; off > 0; off >>= 1) {
    if (tid < off) s_fred[tid] = fmaxf(s_fred[tid], s_fred[tid + off]);
    __syncthreads();
  }
  const float dmax = s_fred[0];
  __syncthreads();

#pragma unroll
  for (int s = 0; s < 8; ++s) {
    const int j = tid + s * 256;
    const float adj = dm[s] + (1.f - m2[s]) * dmax;
    s_key[j] = ((unsigned long long)__float_as_uint(adj) << 32) | (unsigned)j;
  }
  __syncthreads();

  for (int k = 0; k < KNB; ++k) {
    unsigned long long best = 0xFFFFFFFFFFFFFFFFull;
#pragma unroll
    for (int s = 0; s < 8; ++s) {
      const unsigned long long v = s_key[tid + s * 256];
      best = (v < best) ? v : best;
    }
    s_red[tid] = best; __syncthreads();
    for (int off = 128; off > 0; off >>= 1) {
      if (tid < off) {
        const unsigned long long o = s_red[tid + off];
        if (o < s_red[tid]) s_red[tid] = o;
      }
      __syncthreads();
    }
    if (tid == 0) {
      const unsigned long long w = s_red[0];
      const int jwin = (int)(unsigned)(w & 0xFFFFFFFFu);
      const float val = __uint_as_float((unsigned)(w >> 32));
      d_nb[row * KNB + k]   = val;
      e_idx[row * KNB + k]  = jwin;
      e_idx_f[row * KNB + k] = (float)jwin;
      s_key[jwin] = 0xFFFFFFFFFFFFFFFFull;
    }
    __syncthreads();
  }
}

// ---------------------------------------------------------------------------
// Kernel 2: per-residue orthonormal frame (e1,e2,e3) -> 9 floats.
// ---------------------------------------------------------------------------
__global__ __launch_bounds__(256) void pf_frames_kernel(
    const float* __restrict__ X, float* __restrict__ frames) {
  const int idx = blockIdx.x * 256 + threadIdx.x;
  if (idx >= BB * LL) return;
  const float* p = X + (size_t)idx * 12;
  const float v1x = p[0] - p[3], v1y = p[1] - p[4], v1z = p[2] - p[5]; // N - Ca
  const float v2x = p[6] - p[3], v2y = p[7] - p[4], v2z = p[8] - p[5]; // C - Ca
  const float n1 = sqrtf(v1x * v1x + v1y * v1y + v1z * v1z);
  const float i1 = 1.f / (n1 + 1e-6f);
  const float e1x = v1x * i1, e1y = v1y * i1, e1z = v1z * i1;
  const float dp = e1x * v2x + e1y * v2y + e1z * v2z;
  const float u2x = v2x - dp * e1x, u2y = v2y - dp * e1y, u2z = v2z - dp * e1z;
  const float n2 = sqrtf(u2x * u2x + u2y * u2y + u2z * u2z);
  const float i2 = 1.f / (n2 + 1e-6f);
  const float e2x = u2x * i2, e2y = u2y * i2, e2z = u2z * i2;
  const float e3x = e1y * e2z - e1z * e2y;
  const float e3y = e1z * e2x - e1x * e2z;
  const float e3z = e1x * e2y - e1y * e2x;
  float* f = frames + (size_t)idx * 9;
  f[0] = e1x; f[1] = e1y; f[2] = e1z;
  f[3] = e2x; f[4] = e2y; f[5] = e2z;
  f[6] = e3x; f[7] = e3y; f[8] = e3z;
}

// ---------------------------------------------------------------------------
// Kernel 3: node AD features -> (3->128) + LayerNorm.  1 block (128 thr)/(b,l)
// ---------------------------------------------------------------------------
__global__ __launch_bounds__(128) void pf_node_kernel(
    const float* __restrict__ X, const float* __restrict__ Wn,
    const float* __restrict__ bn, const float* __restrict__ g,
    const float* __restrict__ beta, float* __restrict__ V) {
  const int row = blockIdx.x;
  const int l = row % LL;
  const int tid = threadIdx.x;
  __shared__ float ad[3];
  __shared__ float s_sum[128];
  __shared__ float s_sq[128];

  if (tid == 0) {
    float a0 = 0.f, a1 = 0.f, a2 = 0.f;
    if (l >= 1 && l <= LL - 3) {
      const float* Ca = X + 3; // atom index 1
      float P[4][3];
#pragma unroll
      for (int t = 0; t < 4; ++t)
#pragma unroll
        for (int c = 0; c < 3; ++c)
          P[t][c] = Ca[(size_t)(row - 1 + t) * 12 + c];
      float U[3][3];
#pragma unroll
      for (int t = 0; t < 3; ++t) {
        float dx = P[t + 1][0] - P[t][0];
        float dy = P[t + 1][1] - P[t][1];
        float dz = P[t + 1][2] - P[t][2];
        float n = sqrtf(dx * dx + dy * dy + dz * dz);
        float inv = 1.f / fmaxf(n, 1e-12f);
        U[t][0] = dx * inv; U[t][1] = dy * inv; U[t][2] = dz * inv;
      }
      // u2 = U[0], u1 = U[1], u0 = U[2]
      float c21x = U[0][1] * U[1][2] - U[0][2] * U[1][1];
      float c21y = U[0][2] * U[1][0] - U[0][0] * U[1][2];
      float c21z = U[0][0] * U[1][1] - U[0][1] * U[1][0];
      float n21 = sqrtf(c21x * c21x + c21y * c21y + c21z * c21z);
      float i21 = 1.f / fmaxf(n21, 1e-12f);
      c21x *= i21; c21y *= i21; c21z *= i21;
      float c10x = U[1][1] * U[2][2] - U[1][2] * U[2][1];
      float c10y = U[1][2] * U[2][0] - U[1][0] * U[2][2];
      float c10z = U[1][0] * U[2][1] - U[1][1] * U[2][0];
      float n10 = sqrtf(c10x * c10x + c10y * c10y + c10z * c10z);
      float i10 = 1.f / fmaxf(n10, 1e-12f);
      c10x *= i10; c10y *= i10; c10z *= i10;
      float cosA = -(U[1][0] * U[2][0] + U[1][1] * U[2][1] + U[1][2] * U[2][2]);
      cosA = fminf(fmaxf(cosA, -1.f + 1e-6f), 1.f - 1e-6f);
      float A = acosf(cosA);
      float cosD = c21x * c10x + c21y * c10y + c21z * c10z;
      cosD = fminf(fmaxf(cosD, -1.f + 1e-6f), 1.f - 1e-6f);
      float sg = sgnf(U[0][0] * c10x + U[0][1] * c10y + U[0][2] * c10z);
      float D = sg * acosf(cosD);
      a0 = cosf(A); a1 = sinf(A) * cosf(D); a2 = sinf(A) * sinf(D);
    }
    ad[0] = a0; ad[1] = a1; ad[2] = a2;
  }
  __syncthreads();

  const float y = ad[0] * Wn[tid] + ad[1] * Wn[128 + tid] + ad[2] * Wn[256 + tid] + bn[tid];
  s_sum[tid] = y; s_sq[tid] = y * y; __syncthreads();
  for (int off = 64; off > 0; off >>= 1) {
    if (tid < off) { s_sum[tid] += s_sum[tid + off]; s_sq[tid] += s_sq[tid + off]; }
    __syncthreads();
  }
  const float mu = s_sum[0] * (1.f / 128.f);
  const float var = s_sq[0] * (1.f / 128.f) - mu * mu;
  V[(size_t)row * 128 + tid] = (y - mu) * rsqrtf(var + 1e-5f) * g[tid] + beta[tid];
}

// ---------------------------------------------------------------------------
// Kernel 4: W_edge (55x128 f32) -> padded (64x128 f16) for WMMA B operand.
// ---------------------------------------------------------------------------
__global__ __launch_bounds__(256) void pf_wconv_kernel(
    const float* __restrict__ We, _Float16* __restrict__ w16) {
  const int i = blockIdx.x * 256 + threadIdx.x;
  if (i >= EINP * NF) return;
  const int k = i / NF, c = i % NF;
  w16[i] = (k < EIN) ? (_Float16)We[k * NF + c] : (_Float16)0.f;
}

// ---------------------------------------------------------------------------
// Kernel 5: per (b,i): build 32x64 f16 edge-feature tile, WMMA 55->128,
// bias + LayerNorm, write E.  256 threads (8 waves), each wave owns one
// 16-col N tile and both 16-row M tiles; 2 K steps of 32.
// ---------------------------------------------------------------------------
__global__ __launch_bounds__(256) void pf_edge_kernel(
    const float* __restrict__ X, const int* __restrict__ e_idx,
    const float* __restrict__ d_nb, const float* __restrict__ frames,
    const int* __restrict__ residue_idx, const int* __restrict__ chain_labels,
    const float* __restrict__ W_pe, const float* __restrict__ b_pe,
    const float* __restrict__ W_ch, const float* __restrict__ b_ch,
    const _Float16* __restrict__ w16, const float* __restrict__ b_edge,
    const float* __restrict__ g_e, const float* __restrict__ bn_e,
    float* __restrict__ E) {
  const int row = blockIdx.x;        // b*L + i
  const int b   = row / LL;
  const int tid = threadIdx.x;

  __shared__ _Float16 s_feats[32 * EINP];
  __shared__ float    s_out[32 * NF];
  __shared__ float    s_psum[32 * 8];
  __shared__ float    s_psq[32 * 8];
  __shared__ float    s_stats[32 * 2];

  // ---- feature build (wave 0: one thread per neighbor row) ----
  if (tid < 32) {
    const int k = tid;
    _Float16* f = &s_feats[k * EINP];
    if (k < KNB) {
      const int j  = e_idx[row * KNB + k];
      const int gj = b * LL + j;
      const float D = d_nb[row * KNB + k];
      // positional (16)
      int d = residue_idx[gj] - residue_idx[row] + 32;
      d = d < 0 ? 0 : (d > 64 ? 64 : d);
      for (int c = 0; c < 16; ++c) f[c] = (_Float16)(W_pe[d * 16 + c] + b_pe[c]);
      // RBF (16)
      for (int c = 0; c < 16; ++c) {
        const float mu = 2.f + (20.f / 15.f) * (float)c;
        const float t = (D - mu) * (1.f / 1.25f);
        f[16 + c] = (_Float16)__expf(-t * t);
      }
      // orientation (7): dU then quaternion of R = O_i^T O_j
      const float* Oi = frames + (size_t)row * 9;
      const float* Oj = frames + (size_t)gj * 9;
      const float dx = X[(gj * 4 + 1) * 3 + 0] - X[(row * 4 + 1) * 3 + 0];
      const float dy = X[(gj * 4 + 1) * 3 + 1] - X[(row * 4 + 1) * 3 + 1];
      const float dz = X[(gj * 4 + 1) * 3 + 2] - X[(row * 4 + 1) * 3 + 2];
      // einsum('blij,blkj->blki'): du_c = e1[c]*dx + e2[c]*dy + e3[c]*dz
      const float du0 = Oi[0] * dx + Oi[3] * dy + Oi[6] * dz;
      const float du1 = Oi[1] * dx + Oi[4] * dy + Oi[7] * dz;
      const float du2 = Oi[2] * dx + Oi[5] * dy + Oi[8] * dz;
      const float dn = sqrtf(du0 * du0 + du1 * du1 + du2 * du2);
      const float di = 1.f / fmaxf(dn, 1e-12f);
      f[32] = (_Float16)(du0 * di);
      f[33] = (_Float16)(du1 * di);
      f[34] = (_Float16)(du2 * di);
      float R[3][3];
#pragma unroll
      for (int ii = 0; ii < 3; ++ii)
#pragma unroll
        for (int mm = 0; mm < 3; ++mm)
          R[ii][mm] = Oi[ii * 3 + 0] * Oj[mm * 3 + 0] +
                      Oi[ii * 3 + 1] * Oj[mm * 3 + 1] +
                      Oi[ii * 3 + 2] * Oj[mm * 3 + 2];
      const float Rxx = R[0][0], Ryy = R[1][1], Rzz = R[2][2];
      const float m0 = 0.5f * sqrtf(fabsf(1.f + Rxx - Ryy - Rzz));
      const float m1 = 0.5f * sqrtf(fabsf(1.f - Rxx + Ryy - Rzz));
      const float m2 = 0.5f * sqrtf(fabsf(1.f - Rxx - Ryy + Rzz));
      const float qx = sgnf(R[2][1] - R[1][2]) * m0;
      const float qy = sgnf(R[0][2] - R[2][0]) * m1;
      const float qz = sgnf(R[1][0] - R[0][1]) * m2;
      const float tr = 1.f + Rxx + Ryy + Rzz;
      const float qw = 0.5f * sqrtf(fmaxf(tr, 0.f));
      const float qn = sqrtf(qx * qx + qy * qy + qz * qz + qw * qw);
      const float qi = 1.f / fmaxf(qn, 1e-12f);
      f[35] = (_Float16)(qx * qi);
      f[36] = (_Float16)(qy * qi);
      f[37] = (_Float16)(qz * qi);
      f[38] = (_Float16)(qw * qi);
      // chains (16)
      const float same = (chain_labels[gj] == chain_labels[row]) ? 1.f : 0.f;
      for (int c = 0; c < 16; ++c)
        f[39 + c] = (_Float16)(same * W_ch[c] + (1.f - same) * W_ch[16 + c] + b_ch[c]);
      for (int c = EIN; c < EINP; ++c) f[c] = (_Float16)0.f;
    } else {
      for (int c = 0; c < EINP; ++c) f[c] = (_Float16)0.f;
    }
  }
  __syncthreads();

  // ---- WMMA: D(32x128) = A(32x64) * B(64x128) ----
  const int wv = tid >> 5;
  const int lane = tid & 31;
  const int laneLo = lane & 15;
  const int hi = lane >> 4;        // 0 or 1
  const int nb = wv * 16;          // N tile base

  union AF { v16h v; _Float16 h[16]; };
  union CF { v8f v; float f[8]; };
  CF acc0, acc1;
#pragma unroll
  for (int r = 0; r < 8; ++r) { acc0.f[r] = 0.f; acc1.f[r] = 0.f; }

#pragma unroll
  for (int kt = 0; kt < 2; ++kt) {
    AF a0, a1, bf;
#pragma unroll
    for (int e = 0; e < 16; ++e) {
      const int v = e >> 1, h = e & 1;
      // A 16x32 f16 layout (ISA 7.12.2): lanes0-15 K pairs 0..7 then 16..23;
      // lanes16-31 shifted by +8.
      const int ka = kt * 32 + (v & 3) * 2 + (v >> 2) * 16 + hi * 8 + h;
      a0.h[e] = s_feats[laneLo * EINP + ka];
      a1.h[e] = s_feats[(16 + laneLo) * EINP + ka];
      // B 32x16 f16 layout: same K order as A (lanes0-15 K 0..7,16..23; lanes16-31 +8).
      const int kb = kt * 32 + (v & 3) * 2 + (v >> 2) * 16 + hi * 8 + h;
      bf.h[e] = w16[kb * NF + nb + laneLo];
    }
    acc0.v = __builtin_amdgcn_wmma_f32_16x16x32_f16(
        false, a0.v, false, bf.v, (short)0, acc0.v, false, false);
    acc1.v = __builtin_amdgcn_wmma_f32_16x16x32_f16(
        false, a1.v, false, bf.v, (short)0, acc1.v, false, false);
  }

  // C/D layout: VGPR r -> M = r + 8*hi, N = laneLo (+ tile offsets)
#pragma unroll
  for (int r = 0; r < 8; ++r) {
    const int m = r + hi * 8;
    const int n = nb + laneLo;
    s_out[m * NF + n]        = acc0.f[r] + b_edge[n];
    s_out[(16 + m) * NF + n] = acc1.f[r] + b_edge[n];
  }
  __syncthreads();

  // ---- LayerNorm per row (8 threads/row x 16 cols) + write ----
  const int rrow = tid >> 3;
  const int sub = tid & 7;
  float ps = 0.f, pq = 0.f;
  for (int c = sub * 16; c < sub * 16 + 16; ++c) {
    const float v = s_out[rrow * NF + c];
    ps += v; pq += v * v;
  }
  s_psum[rrow * 8 + sub] = ps; s_psq[rrow * 8 + sub] = pq;
  __syncthreads();
  if (sub == 0) {
    float S = 0.f, Q = 0.f;
    for (int u = 0; u < 8; ++u) { S += s_psum[rrow * 8 + u]; Q += s_psq[rrow * 8 + u]; }
    const float mu = S * (1.f / 128.f);
    const float var = Q * (1.f / 128.f) - mu * mu;
    s_stats[rrow * 2 + 0] = mu;
    s_stats[rrow * 2 + 1] = rsqrtf(var + 1e-5f);
  }
  __syncthreads();
  if (rrow < KNB) {
    const float mu = s_stats[rrow * 2 + 0];
    const float rs = s_stats[rrow * 2 + 1];
    float* Eo = E + ((size_t)row * KNB + rrow) * NF;
    for (int c = sub * 16; c < sub * 16 + 16; ++c)
      Eo[c] = (s_out[rrow * NF + c] - mu) * rs * g_e[c] + bn_e[c];
  }
}

// ---------------------------------------------------------------------------
extern "C" void kernel_launch(void* const* d_in, const int* in_sizes, int n_in,
                              void* d_out, int out_size, void* d_ws, size_t ws_size,
                              hipStream_t stream) {
  (void)in_sizes; (void)n_in; (void)out_size; (void)ws_size;
  const float* X            = (const float*)d_in[0];
  const float* mask         = (const float*)d_in[1];
  const int*   residue_idx  = (const int*)d_in[2];
  const int*   chain_labels = (const int*)d_in[3];
  const float* W_pe   = (const float*)d_in[4];
  const float* b_pe   = (const float*)d_in[5];
  const float* W_ch   = (const float*)d_in[6];
  const float* b_ch   = (const float*)d_in[7];
  const float* W_node = (const float*)d_in[8];
  const float* b_node = (const float*)d_in[9];
  const float* W_edge = (const float*)d_in[10];
  const float* b_edge = (const float*)d_in[11];
  const float* g_nodes  = (const float*)d_in[12];
  const float* bn_nodes = (const float*)d_in[13];
  const float* g_edges  = (const float*)d_in[14];
  const float* bn_edges = (const float*)d_in[15];

  float* out   = (float*)d_out;
  float* V     = out;                                        // B*L*128
  float* E     = out + (size_t)BB * LL * NF;                 // B*L*30*128
  float* Eidxf = E + (size_t)BB * LL * KNB * NF;             // B*L*30

  char* ws = (char*)d_ws;
  constexpr size_t SZ_EIDX   = (size_t)BB * LL * KNB * sizeof(int);    // 983040
  constexpr size_t SZ_DNB    = (size_t)BB * LL * KNB * sizeof(float);  // 983040
  constexpr size_t SZ_FRAMES = (size_t)BB * LL * 9 * sizeof(float);    // 294912
  int*      e_idx  = (int*)ws;
  float*    d_nb   = (float*)(ws + SZ_EIDX);
  float*    frames = (float*)(ws + SZ_EIDX + SZ_DNB);
  _Float16* w16    = (_Float16*)(ws + SZ_EIDX + SZ_DNB + SZ_FRAMES);

  pf_topk_kernel<<<BB * LL, 256, 0, stream>>>(X, mask, d_nb, e_idx, Eidxf);
  pf_frames_kernel<<<(BB * LL + 255) / 256, 256, 0, stream>>>(X, frames);
  pf_wconv_kernel<<<(EINP * NF + 255) / 256, 256, 0, stream>>>(W_edge, w16);
  pf_node_kernel<<<BB * LL, 128, 0, stream>>>(X, W_node, b_node, g_nodes, bn_nodes, V);
  pf_edge_kernel<<<BB * LL, 256, 0, stream>>>(X, e_idx, d_nb, frames,
                                              residue_idx, chain_labels,
                                              W_pe, b_pe, W_ch, b_ch,
                                              w16, b_edge, g_edges, bn_edges, E);
}
